// LSTMRegressor_57037165691479
// MI455X (gfx1250) — matrix operand order repaired and checked by reading, not compiled, not run on hardware
//
#include <hip/hip_runtime.h>

typedef __attribute__((ext_vector_type(16))) _Float16 v16h;
typedef __attribute__((ext_vector_type(8)))  _Float16 v8h;
typedef __attribute__((ext_vector_type(8)))  float    v8f;

#define BATCH 2048
#define SEQ   512
#define HID   51
#define GATES 204          // 4*HID
#define NT    13           // N tiles of 16 (208 padded gate cols)
#define GROW  209          // gates LDS row stride (odd -> bank-conflict free)
#define AROW  72           // hA row stride in halfs (144B rows, 16B aligned)

__device__ __forceinline__ float sigmoidf_fast(float x) {
    // 1 / (1 + 2^(-x*log2(e)))  -> v_exp_f32 + v_rcp_f32
    float e = __builtin_amdgcn_exp2f(-1.4426950408889634f * x);
    return __builtin_amdgcn_rcpf(1.0f + e);
}
__device__ __forceinline__ float tanhf_fast(float x) {
    // tanh(x) = 2*sigmoid(2x) - 1
    float e = __builtin_amdgcn_exp2f(-2.8853900817779268f * x);
    return __builtin_amdgcn_rcpf(1.0f + e) * 2.0f - 1.0f;
}

__global__ __launch_bounds__(128, 4)
void lstm_fused_wmma(const float* __restrict__ x,     const float* __restrict__ W_ih,
                     const float* __restrict__ W_hh,  const float* __restrict__ b_ih,
                     const float* __restrict__ b_hh,  const float* __restrict__ W_out,
                     const float* __restrict__ b_out, float* __restrict__ out)
{
    __shared__ alignas(16) _Float16 hA[16][AROW];   // A staging: [m][k], k = h(0..50) | x(51) | 1(52) | pad
    __shared__ float gates[16][GROW];               // raw gate pre-activations

    const int tid = threadIdx.x;
    const int wl  = tid & 31;                                   // lane in wave
    const int swv = __builtin_amdgcn_readfirstlane(tid) >> 5;   // wave id as SGPR (scalar!)
    const int nlo = wl & 15;
    const int hi  = wl >> 4;
    const int m0  = blockIdx.x * 16;                            // batch-row base

    // ---- Pack augmented weight matrix B into registers (WMMA f16 B layout).
    // B element e (0..15) of lane: k = c*32 + (e&8)*2 + hi*8 + (e&7), n = wl&15.
    // Rows: k<51 -> W_hh[g][k]; k==51 -> W_ih[g]; k==52 -> b_ih[g]+b_hh[g]; else 0.
    v16h Bt[4][2];
    #pragma unroll
    for (int q = 0; q < 4; ++q) {
        const int t = swv + 4 * q;          // scalar tile index
        #pragma unroll
        for (int c = 0; c < 2; ++c) {
            v16h bb;
            #pragma unroll
            for (int e = 0; e < 16; ++e) {
                const int k = c * 32 + (e & 8) * 2 + hi * 8 + (e & 7);
                const int g = t * 16 + nlo;
                float w = 0.0f;
                if (t < NT && g < GATES) {
                    if (k < HID)            w = W_hh[g * HID + k];
                    else if (k == HID)      w = W_ih[g];
                    else if (k == HID + 1)  w = b_ih[g] + b_hh[g];
                }
                bb[e] = (_Float16)w;
            }
            Bt[q][c] = bb;
        }
    }

    // ---- Init A staging: zeros (h0 = 0, pads = 0), constant-1 column for bias.
    for (int i = tid; i < 16 * AROW; i += 128) ((_Float16*)hA)[i] = (_Float16)0.0f;
    __syncthreads();
    if (tid < 16) hA[tid][HID + 1] = (_Float16)1.0f;
    __syncthreads();

    // ---- Activation assignment: each WAVE owns 4 whole batch rows; 8 lanes per row.
    const int rowm = 4 * swv + (wl & 3);             // this lane's batch row (0..15)
    const int sub  = wl >> 2;                        // 0..7 hidden-slice id within the row
    const int jcnt  = (sub < 3) ? 7 : 6;             // 3*7 + 5*6 = 51
    const int jbase = (sub < 3) ? sub * 7 : 21 + (sub - 3) * 6;

    float cst[7];    // register-resident cell state slice
    float wo[7];
    #pragma unroll
    for (int q = 0; q < 7; ++q) {
        cst[q] = 0.0f;
        wo[q]  = (q < jcnt) ? W_out[jbase + q] : 0.0f;
    }
    const float bout = b_out[0];

    const float* xrow = x   + (size_t)(m0 + rowm) * SEQ;
    float*       orow = out + (size_t)(m0 + rowm) * SEQ;

    float xt = (wl < 4) ? xrow[0] : 0.0f;            // lanes 0-3 of each wave feed 4 rows

    for (int t = 0; t < SEQ; ++t) {
        // Stage x_t into augmented A column; prefetch next x early.
        if (wl < 4) hA[rowm][HID] = (_Float16)xt;
        const float xn = (wl < 4 && (t + 1) < SEQ) ? xrow[t + 1] : 0.0f;
        __syncthreads();

        // Load A tile (ISA 16-bit A layout): row = wl&15, K runs {hi*8..+7, +16..} per 32-K chunk.
        v16h A0, A1;
        {
            const int koff = hi * 8;
            v8h a0l = *(const v8h*)&hA[nlo][koff];
            v8h a0h = *(const v8h*)&hA[nlo][koff + 16];
            v8h a1l = *(const v8h*)&hA[nlo][koff + 32];
            v8h a1h = *(const v8h*)&hA[nlo][koff + 48];
            #pragma unroll
            for (int e = 0; e < 8; ++e) {
                A0[e] = a0l[e]; A0[e + 8] = a0h[e];
                A1[e] = a1l[e]; A1[e + 8] = a1h[e];
            }
        }

        // Gate GEMM: this wave's N-tiles (scalar guard -> no EXEC games around WMMA).
        const int mhi = 8 * hi;
        #pragma unroll
        for (int q = 0; q < 4; ++q) {
            const int tt = swv + 4 * q;              // scalar
            if (tt < NT) {
                v8f acc = {0.f, 0.f, 0.f, 0.f, 0.f, 0.f, 0.f, 0.f};
                acc = __builtin_amdgcn_wmma_f32_16x16x32_f16(
                        false, A0, false, Bt[q][0], (short)0, acc, false, false);
                acc = __builtin_amdgcn_wmma_f32_16x16x32_f16(
                        false, A1, false, Bt[q][1], (short)0, acc, false, false);
                #pragma unroll
                for (int r = 0; r < 8; ++r)
                    gates[mhi + r][tt * 16 + nlo] = acc[r];   // C layout: M=r+8*hi, N=lane%16
            }
        }
        __syncthreads();

        // Activations + state update on this lane's (rowm, j) slice.
        float part = 0.0f;
        #pragma unroll
        for (int q = 0; q < 7; ++q) {
            if (q < jcnt) {
                const int j = jbase + q;
                const float iv = sigmoidf_fast(gates[rowm][j]);
                const float fv = sigmoidf_fast(gates[rowm][j + HID]);
                const float gv = tanhf_fast  (gates[rowm][j + 2 * HID]);
                const float ov = sigmoidf_fast(gates[rowm][j + 3 * HID]);
                const float cn = fv * cst[q] + iv * gv;
                cst[q] = cn;
                const float hn = ov * tanhf_fast(cn);
                hA[rowm][j] = (_Float16)hn;          // feed next step's A tile
                part += hn * wo[q];
            }
        }

        // out[b][t] = h . W_out + b_out : all 8 slices of a row live in this wave
        // (lanes rowlane + {0,4,8,...,28}); xor-reduce over lane bits 2..4.
        part += __shfl_xor(part, 4, 32);
        part += __shfl_xor(part, 8, 32);
        part += __shfl_xor(part, 16, 32);
        if (wl < 4) orow[t] = part + bout;

        xt = xn;
        // No barrier needed here: next iteration's top barrier orders the
        // h/x writes against A reads, and gate writes happen after it.
    }
}

extern "C" void kernel_launch(void* const* d_in, const int* in_sizes, int n_in,
                              void* d_out, int out_size, void* d_ws, size_t ws_size,
                              hipStream_t stream) {
    const float* x     = (const float*)d_in[0];
    const float* W_ih  = (const float*)d_in[1];
    const float* W_hh  = (const float*)d_in[2];
    const float* b_ih  = (const float*)d_in[3];
    const float* b_hh  = (const float*)d_in[4];
    const float* W_out = (const float*)d_in[5];
    const float* b_out = (const float*)d_in[6];
    float* out = (float*)d_out;

    dim3 grid(BATCH / 16);   // 128 blocks x 16 batch rows
    dim3 block(128);         // 4 waves (wave32)
    lstm_fused_wmma<<<grid, block, 0, stream>>>(x, W_ih, W_hh, b_ih, b_hh, W_out, b_out, out);
}
